// BB_24404004176378
// MI455X (gfx1250) — matrix operand order repaired and checked by reading, not compiled, not run on hardware
//
#include <hip/hip_runtime.h>

typedef _Float16 f16;
typedef __attribute__((ext_vector_type(16))) _Float16 v16h;
typedef __attribute__((ext_vector_type(8)))  float    v8f;
typedef __attribute__((ext_vector_type(4)))  unsigned u32x4;
typedef __attribute__((ext_vector_type(2)))  unsigned u32x2;
typedef __attribute__((ext_vector_type(4)))  int      v4i;

union Frag {
  v16h v;
  f16 h[16];
  unsigned u[8];
  u32x4 q4[2];
};

union Frag4 {
  u32x2 q2;
  f16 h[4];
};

#define B_ 4
#define C_ 32
#define H_ 64
#define W_ 64
#define HEADS_ 8
#define SF_ 4
#define D_ 288
#define E_ 144
#define DH_ 18
#define N_ 4096
#define NCH_ 1024
#define EP_ 160          // E padded to multiple of 32
#define DP_ 32           // head dim padded 18 -> 32
#define FQKV_ 432
#define SCALE_ 0.23570226039551584f   // 18^-0.5

#if __has_builtin(__builtin_amdgcn_global_load_async_to_lds_b128)
#define HAVE_ASYNC_LDS 1
typedef __attribute__((address_space(1))) v4i* gv4i_t;
typedef __attribute__((address_space(3))) v4i* lv4i_t;
#else
#define HAVE_ASYNC_LDS 0
#endif

static __device__ __forceinline__ v8f wmma16x16x32(v16h a, v16h b, v8f c) {
  return __builtin_amdgcn_wmma_f32_16x16x32_f16(false, a, false, b, (short)0, c, false, false);
}

// ---------------------------------------------------------------- utilities

__global__ __launch_bounds__(256) void fill0_kernel(unsigned* p, int nwords) {
  int i = blockIdx.x * 256 + threadIdx.x;
  if (i < nwords) p[i] = 0u;
}

// dst (Cc x R), dst[c*R + r] = src[r*Cc + c], converted to f16
__global__ __launch_bounds__(256) void transpose_f16_kernel(const float* __restrict__ src,
                                                            f16* __restrict__ dst,
                                                            int R, int Cc) {
  int i = blockIdx.x * 256 + threadIdx.x;
  if (i >= R * Cc) return;
  int r = i / Cc, c = i - r * Cc;
  dst[(size_t)c * R + r] = (f16)src[i];
}

// cols[b, d, n] = padded(src)[b, c, h+i, w+j], d = c*9 + i*3 + j, n = h*W + w
__global__ __launch_bounds__(256) void unfold_kernel(const float* __restrict__ src,
                                                     float* __restrict__ cols) {
  int idx = blockIdx.x * 256 + threadIdx.x;
  if (idx >= B_ * D_ * N_) return;
  int n = idx % N_;
  int rest = idx / N_;
  int d = rest % D_;
  int b = rest / D_;
  int c = d / 9;
  int ij = d - c * 9;
  int i = ij / 3, j = ij - (ij / 3) * 3;
  int h = n / W_, w = n - (n / W_) * W_;
  int sh = h + i - 1, sw = w + j - 1;
  float v = 0.0f;
  if (sh >= 0 && sh < H_ && sw >= 0 && sw < W_)
    v = src[(((size_t)b * C_ + c) * H_ + sh) * W_ + sw];
  cols[idx] = v;
}

// LayerNorm over D per token, output token-major f16 (B, N, D)
__global__ __launch_bounds__(256) void ln_kernel(const float* __restrict__ cols,
                                                 const float* __restrict__ g,
                                                 const float* __restrict__ bt,
                                                 f16* __restrict__ colsn) {
  int t = blockIdx.x * 256 + threadIdx.x;
  if (t >= B_ * N_) return;
  int b = t / N_, n = t - b * N_;
  const float* base = cols + (size_t)b * D_ * N_ + n;
  float s = 0.0f, s2 = 0.0f;
  for (int d = 0; d < D_; d++) {
    float v = base[(size_t)d * N_];
    s += v; s2 += v * v;
  }
  float mu = s * (1.0f / D_);
  float var = s2 * (1.0f / D_) - mu * mu;
  float inv = rsqrtf(var + 1e-5f);
  f16* dst = colsn + (size_t)t * D_;
  for (int d = 0; d < D_; d++) {
    float v = (base[(size_t)d * N_] - mu) * inv * g[d] + bt[d];
    dst[d] = (f16)v;
  }
}

// t_img[b,c,h,w] = sum_{i,j valid} cols2[b, c*9+i*3+j, (h+1-i)*W + (w+1-j)]
__global__ __launch_bounds__(256) void fold_kernel(const float* __restrict__ cols2,
                                                   float* __restrict__ t_img) {
  int idx = blockIdx.x * 256 + threadIdx.x;
  if (idx >= B_ * C_ * H_ * W_) return;
  int w = idx % W_;
  int rest = idx / W_;
  int h = rest % H_;
  rest /= H_;
  int c = rest % C_;
  int b = rest / C_;
  float acc = 0.0f;
  #pragma unroll
  for (int i = 0; i < 3; i++) {
    int sh = h + 1 - i;
    if (sh < 0 || sh >= H_) continue;
    #pragma unroll
    for (int j = 0; j < 3; j++) {
      int sw = w + 1 - j;
      if (sw < 0 || sw >= W_) continue;
      acc += cols2[((size_t)b * D_ + (c * 9 + i * 3 + j)) * N_ + sh * W_ + sw];
    }
  }
  t_img[idx] = acc;
}

// out = x + elu(conv3x3(t_img) + conv_b)
__global__ __launch_bounds__(256) void conv_elu_kernel(const float* __restrict__ x,
                                                       const float* __restrict__ t_img,
                                                       const float* __restrict__ cw,
                                                       const float* __restrict__ cb,
                                                       float* __restrict__ out) {
  int idx = blockIdx.x * 256 + threadIdx.x;
  if (idx >= B_ * C_ * H_ * W_) return;
  int w = idx % W_;
  int rest = idx / W_;
  int h = rest % H_;
  rest /= H_;
  int co = rest % C_;
  int b = rest / C_;
  float acc = cb[co];
  for (int ci = 0; ci < C_; ci++) {
    const float* tp = t_img + ((size_t)b * C_ + ci) * H_ * W_;
    const float* wp = cw + ((size_t)co * C_ + ci) * 9;
    #pragma unroll
    for (int i = 0; i < 3; i++) {
      int sh = h + i - 1;
      if (sh < 0 || sh >= H_) continue;
      #pragma unroll
      for (int j = 0; j < 3; j++) {
        int sw = w + j - 1;
        if (sw < 0 || sw >= W_) continue;
        acc += tp[sh * W_ + sw] * wp[i * 3 + j];
      }
    }
  }
  float e = acc > 0.0f ? acc : (__expf(acc) - 1.0f);
  out[idx] = x[idx] + e;
}

// ------------------------------------------------------------- dense GEMM
// out[n, f] = sum_k A[n,k] * Bm[k,f]  (per batch in grid.z)
// 64x64 block tile, 8 waves, each wave: 1 row-tile x 2 col-tiles of 16x16.
// mode 0: out_t[(b*N+n)*EP + f] = acc + bias[f]                  (f16)
// mode 1: scatter to q/k/v padded (B,H,SF,1024,32)               (f16)
// mode 2: cols2[(b*D+f)*N + n] = acc + bias[f] + colsres[...]    (f32)
__global__ __launch_bounds__(256) void gemm_wmma_kernel(const f16* __restrict__ A, int lda,
                                                        const f16* __restrict__ Bm, int ldb,
                                                        int F, int Kdim, int mode,
                                                        const float* __restrict__ bias,
                                                        f16* __restrict__ out_t,
                                                        f16* __restrict__ qb,
                                                        f16* __restrict__ kb,
                                                        f16* __restrict__ vb,
                                                        const float* __restrict__ colsres,
                                                        float* __restrict__ cols2) {
  __shared__ __align__(16) f16 As[64 * 32];    // (row, k) row-major
  __shared__ __align__(16) f16 Bst[64 * 32];   // transposed: (col, k)
  int tid = threadIdx.x;
  int lane = tid & 31, wave = tid >> 5;
  int m = lane & 15, hs = lane >> 4;
  int b = blockIdx.z;
  int rowbase = blockIdx.y * 64;
  int colbase = blockIdx.x * 64;
  const f16* Ab = A + (size_t)b * N_ * lda;

  int rt = wave & 3;        // row tile 0..3
  int ctp = wave >> 2;      // col tile pair 0..1
  v8f acc[2] = {};

  for (int k0 = 0; k0 < Kdim; k0 += 32) {
    // stage A (64x32), coalesced 16B per thread
    {
      int r = tid >> 2, sA = (tid & 3) * 8;
      *(u32x4*)&As[r * 32 + sA] =
          *(const u32x4*)&Ab[(size_t)(rowbase + r) * lda + k0 + sA];
    }
    // stage B transposed: read (k,col) row-major, write (col,k)
    {
      int r2 = tid >> 3, s2 = (tid & 7) * 8;
      int col = colbase + s2;
      Frag tmp;
      if (col < F) {
        tmp.q4[0] = *(const u32x4*)&Bm[(size_t)(k0 + r2) * ldb + col];
      } else {
        u32x4 z = {0u, 0u, 0u, 0u};
        tmp.q4[0] = z;
      }
      #pragma unroll
      for (int tt = 0; tt < 8; tt++) Bst[(s2 + tt) * 32 + r2] = tmp.h[tt];
    }
    __syncthreads();

    // A fragment (16x32): row = lane&15, K chunks at 8*hs and 16+8*hs
    Frag af;
    {
      const f16* arow = &As[(rt * 16 + m) * 32 + 8 * hs];
      af.q4[0] = *(const u32x4*)&arow[0];
      af.q4[1] = *(const u32x4*)&arow[16];
    }
    // B fragments (32x16): col = lane&15, K chunks at 8*hs and 16+8*hs
    Frag bf0, bf1;
    {
      const f16* b0 = &Bst[(ctp * 32 + m) * 32 + 8 * hs];
      bf0.q4[0] = *(const u32x4*)&b0[0];
      bf0.q4[1] = *(const u32x4*)&b0[16];
      const f16* b1 = &Bst[(ctp * 32 + 16 + m) * 32 + 8 * hs];
      bf1.q4[0] = *(const u32x4*)&b1[0];
      bf1.q4[1] = *(const u32x4*)&b1[16];
    }
    acc[0] = wmma16x16x32(af.v, bf0.v, acc[0]);
    acc[1] = wmma16x16x32(af.v, bf1.v, acc[1]);
    __syncthreads();
  }

  // write out: C layout row = r + 8*hs, col = lane&15
  #pragma unroll
  for (int j = 0; j < 2; j++) {
    int ct = ctp * 2 + j;
    #pragma unroll
    for (int r = 0; r < 8; r++) {
      int row = rowbase + rt * 16 + r + 8 * hs;
      int col = colbase + ct * 16 + m;
      if (col >= F) continue;
      float val = acc[j][r];
      if (mode == 0) {
        out_t[((size_t)b * N_ + row) * EP_ + col] = (f16)(val + bias[col]);
      } else if (mode == 1) {
        int which = col / E_;
        int e = col - which * E_;
        int hh = e / DH_;
        int dd = e - hh * DH_;
        int s = row >> 10;
        int rr = row & 1023;
        f16* dst = (which == 0) ? qb : ((which == 1) ? kb : vb);
        dst[((((size_t)b * HEADS_ + hh) * SF_ + s) * NCH_ + rr) * DP_ + dd] = (f16)val;
      } else {
        size_t idx = ((size_t)b * D_ + col) * N_ + row;
        cols2[idx] = val + bias[col] + colsres[idx];
      }
    }
  }
}

// ---------------------------------------------------- flash attention
// grid: (8 row-blocks of 128, HEADS*SF, B); each wave: 16 query rows.
// K and V tiles are staged cooperatively into LDS once per 32-key step
// (they are shared by all 8 waves); V is staged transposed (d-major) so
// every fragment load is a contiguous ds_load_b128.
__global__ __launch_bounds__(256) void attn_kernel(const f16* __restrict__ qb,
                                                   const f16* __restrict__ kb,
                                                   const f16* __restrict__ vb,
                                                   f16* __restrict__ ob) {
  __shared__ __align__(16) f16 Ps[8 * 512];   // per-wave 16x32 P scratch
  __shared__ __align__(16) f16 Ks[32 * 32];   // (key, d) direct copy
  __shared__ __align__(16) f16 Vt[32 * 32];   // transposed: (d, key)
  int tid = threadIdx.x;
  int lane = tid & 31, wave = tid >> 5;
  int m = lane & 15, hs = lane >> 4;
  int hsid = blockIdx.y;
  int h = hsid / SF_, s = hsid - h * SF_;
  int b = blockIdx.z;
  int rowbase = blockIdx.x * 128 + wave * 16;   // row within 1024-chunk
  size_t sliceOff = (((size_t)b * HEADS_ + h) * SF_ + s) * (size_t)NCH_ * DP_;
  const f16* q = qb + sliceOff;
  const f16* kM = kb + sliceOff;
  const f16* vM = vb + sliceOff;
  f16* lds = &Ps[wave * 512];

  // Q fragment (A layout), loaded once
  Frag qf;
  {
    const f16* qr = &q[(size_t)(rowbase + m) * DP_];
    qf.q4[0] = *(const u32x4*)&qr[8 * hs];
    qf.q4[1] = *(const u32x4*)&qr[16 + 8 * hs];
  }

  v8f o0 = {}, o1 = {};
  float mrun[8], lrun[8];
  #pragma unroll
  for (int r = 0; r < 8; r++) { mrun[r] = -3.0e38f; lrun[r] = 0.0f; }

  for (int kt = 0; kt < NCH_; kt += 32) {
    // ---- stage K tile: 32x32 halves, fully contiguous in memory ----
    const f16* kglob = kM + (size_t)kt * DP_;
#if HAVE_ASYNC_LDS
    if (tid < 128) {
      void* gsrc = (void*)(kglob + tid * 8);   // drop const, generic ptr
      void* ldst = (void*)&Ks[tid * 8];
      __builtin_amdgcn_global_load_async_to_lds_b128(
          (gv4i_t)gsrc, (lv4i_t)ldst, 0, 0);
    }
#else
    if (tid < 128) {
      *(u32x4*)&Ks[tid * 8] = *(const u32x4*)&kglob[tid * 8];
    }
#endif
    // ---- stage V tile transposed: read (key,d), write (d,key) ----
    {
      const f16* vglob = vM + (size_t)kt * DP_;
      Frag4 tmp;
      int i0 = tid * 4;
      int row = i0 >> 5;          // key within tile
      int c0 = i0 & 31;           // first d
      tmp.q2 = *(const u32x2*)&vglob[i0];
      #pragma unroll
      for (int ii = 0; ii < 4; ii++) Vt[(c0 + ii) * 32 + row] = tmp.h[ii];
    }
#if HAVE_ASYNC_LDS
    asm volatile("s_wait_asynccnt 0x0" ::: "memory");
#endif
    __syncthreads();

    // K fragments (B layout): d chunks at 8*hs and 16+8*hs per lane
    Frag kf0, kf1;
    {
      const f16* kr0 = &Ks[m * 32 + 8 * hs];
      kf0.q4[0] = *(const u32x4*)&kr0[0];
      kf0.q4[1] = *(const u32x4*)&kr0[16];
      const f16* kr1 = &Ks[(16 + m) * 32 + 8 * hs];
      kf1.q4[0] = *(const u32x4*)&kr1[0];
      kf1.q4[1] = *(const u32x4*)&kr1[16];
    }
    v8f zero = {};
    v8f s0 = wmma16x16x32(qf.v, kf0.v, zero);
    v8f s1 = wmma16x16x32(qf.v, kf1.v, zero);

    // streaming softmax over the 32 new columns
    #pragma unroll
    for (int r = 0; r < 8; r++) {
      float a0 = s0[r] * SCALE_;
      float a1 = s1[r] * SCALE_;
      float tmax = fmaxf(a0, a1);
      tmax = fmaxf(tmax, __shfl_xor(tmax, 1));
      tmax = fmaxf(tmax, __shfl_xor(tmax, 2));
      tmax = fmaxf(tmax, __shfl_xor(tmax, 4));
      tmax = fmaxf(tmax, __shfl_xor(tmax, 8));
      float mnew = fmaxf(mrun[r], tmax);
      float sc = __expf(mrun[r] - mnew);
      mrun[r] = mnew;
      float p0 = __expf(a0 - mnew);
      float p1 = __expf(a1 - mnew);
      float rs = p0 + p1;
      rs += __shfl_xor(rs, 1);
      rs += __shfl_xor(rs, 2);
      rs += __shfl_xor(rs, 4);
      rs += __shfl_xor(rs, 8);
      lrun[r] = lrun[r] * sc + rs;
      o0[r] *= sc;
      o1[r] *= sc;
      // C-layout -> LDS (row = r + 8*hs, col = lane&15)
      lds[(r + 8 * hs) * 32 + m]      = (f16)p0;
      lds[(r + 8 * hs) * 32 + 16 + m] = (f16)p1;
    }

    // P as A fragment from LDS (same-wave DS ops are in-order)
    Frag pf;
    {
      const f16* pr = &lds[m * 32 + 8 * hs];
      pf.q4[0] = *(const u32x4*)&pr[0];
      pf.q4[1] = *(const u32x4*)&pr[16];
    }
    // V fragments (B layout) from transposed LDS: key chunks at 8*hs and 16+8*hs
    Frag vf0, vf1;
    {
      const f16* v0 = &Vt[m * 32 + 8 * hs];
      vf0.q4[0] = *(const u32x4*)&v0[0];
      vf0.q4[1] = *(const u32x4*)&v0[16];
      const f16* v1 = &Vt[(16 + m) * 32 + 8 * hs];
      vf1.q4[0] = *(const u32x4*)&v1[0];
      vf1.q4[1] = *(const u32x4*)&v1[16];
    }
    o0 = wmma16x16x32(pf.v, vf0.v, o0);
    o1 = wmma16x16x32(pf.v, vf1.v, o1);
    __syncthreads();
  }

  // normalize and write 18 valid head dims into (B, N, EP) buffer
  #pragma unroll
  for (int r = 0; r < 8; r++) {
    float inv = 1.0f / lrun[r];
    int rowc = rowbase + r + 8 * hs;
    int n = s * NCH_ + rowc;
    size_t obase = ((size_t)b * N_ + n) * EP_ + h * DH_;
    ob[obase + m] = (f16)(o0[r] * inv);
    if (m < 2) ob[obase + 16 + m] = (f16)(o1[r] * inv);
  }
}

// ------------------------------------------------------------------ host

extern "C" void kernel_launch(void* const* d_in, const int* in_sizes, int n_in,
                              void* d_out, int out_size, void* d_ws, size_t ws_size,
                              hipStream_t stream) {
  (void)in_sizes; (void)n_in; (void)out_size; (void)ws_size;
  const float* x      = (const float*)d_in[0];
  const float* ln_g   = (const float*)d_in[1];
  const float* ln_b   = (const float*)d_in[2];
  const float* conv_w = (const float*)d_in[3];
  const float* conv_b = (const float*)d_in[4];
  float* out = (float*)d_out;

  char* ws = (char*)d_ws;
  size_t off = 0;
  float* cols  = (float*)(ws + off); off += (size_t)B_ * D_ * N_ * 4;       // 18.9 MB
  float* cols2 = (float*)(ws + off); off += (size_t)B_ * D_ * N_ * 4;       // 18.9 MB
  float* t_img = (float*)(ws + off); off += (size_t)B_ * C_ * H_ * W_ * 4;  // 2 MB
  f16* colsn   = (f16*)(ws + off);   off += (size_t)B_ * N_ * D_ * 2;       // 9.4 MB
  size_t zoff = off;                                                        // zero region:
  f16* t16  = (f16*)(ws + off); off += (size_t)B_ * N_ * EP_ * 2;           // 5.2 MB
  f16* qb   = (f16*)(ws + off); off += (size_t)B_ * HEADS_ * SF_ * NCH_ * DP_ * 2;
  f16* kb   = (f16*)(ws + off); off += (size_t)B_ * HEADS_ * SF_ * NCH_ * DP_ * 2;
  f16* vb   = (f16*)(ws + off); off += (size_t)B_ * HEADS_ * SF_ * NCH_ * DP_ * 2;
  f16* ob   = (f16*)(ws + off); off += (size_t)B_ * N_ * EP_ * 2;           // 5.2 MB
  f16* wrT   = (f16*)(ws + off); off += (size_t)D_ * E_ * 2;                // (288 x 144)
  f16* wqkvT = (f16*)(ws + off); off += (size_t)EP_ * FQKV_ * 2;            // (160 x 432)
  f16* weT   = (f16*)(ws + off); off += (size_t)EP_ * D_ * 2;               // (160 x 288)
  size_t zwords = (off - zoff) / 4;

  // zero all pad-sensitive buffers once per launch (pads are never rewritten)
  fill0_kernel<<<dim3((unsigned)((zwords + 255) / 256)), dim3(256), 0, stream>>>(
      (unsigned*)(ws + zoff), (int)zwords);

  const float* cur = x;
  for (int L = 0; L < 3; L++) {
    const float* wr   = (const float*)d_in[5 + L * 5 + 0];
    const float* br   = (const float*)d_in[5 + L * 5 + 1];
    const float* wqkv = (const float*)d_in[5 + L * 5 + 2];
    const float* we   = (const float*)d_in[5 + L * 5 + 3];
    const float* be   = (const float*)d_in[5 + L * 5 + 4];

    // weight transposes to f16 (dst[c*R + r] = src[r*Cc + c])
    transpose_f16_kernel<<<dim3((E_ * D_ + 255) / 256), dim3(256), 0, stream>>>(wr, wrT, E_, D_);
    transpose_f16_kernel<<<dim3((FQKV_ * E_ + 255) / 256), dim3(256), 0, stream>>>(wqkv, wqkvT, FQKV_, E_);
    transpose_f16_kernel<<<dim3((D_ * E_ + 255) / 256), dim3(256), 0, stream>>>(we, weT, D_, E_);

    unfold_kernel<<<dim3((B_ * D_ * N_ + 255) / 256), dim3(256), 0, stream>>>(cur, cols);
    ln_kernel<<<dim3((B_ * N_ + 255) / 256), dim3(256), 0, stream>>>(cols, ln_g, ln_b, colsn);

    // GEMM1: t[n,e] = colsn[n,:] . wrT[:,e] + br[e]      (N x 288) @ (288 x 144)
    gemm_wmma_kernel<<<dim3(3, N_ / 64, B_), dim3(256), 0, stream>>>(
        colsn, D_, wrT, E_, E_, D_, 0, br, t16, nullptr, nullptr, nullptr, nullptr, nullptr);

    // GEMM2: qkv[n,f] = t[n,:] . wqkvT[:,f]              (N x 160) @ (160 x 432), scatter q/k/v
    gemm_wmma_kernel<<<dim3(7, N_ / 64, B_), dim3(256), 0, stream>>>(
        t16, EP_, wqkvT, FQKV_, FQKV_, EP_, 1, nullptr, nullptr, qb, kb, vb, nullptr, nullptr);

    // flash attention over 128 (b,h,s) slices
    attn_kernel<<<dim3(NCH_ / 128, HEADS_ * SF_, B_), dim3(256), 0, stream>>>(qb, kb, vb, ob);

    // GEMM4: cols2[f,n] = o[n,:] . weT[:,f] + be[f] + cols[f,n]   (N x 160) @ (160 x 288)
    gemm_wmma_kernel<<<dim3(5, N_ / 64, B_), dim3(256), 0, stream>>>(
        ob, EP_, weT, D_, D_, EP_, 2, be, nullptr, nullptr, nullptr, nullptr, cols, cols2);

    fold_kernel<<<dim3((B_ * C_ * H_ * W_ + 255) / 256), dim3(256), 0, stream>>>(cols2, t_img);
    cur = t_img;
  }

  conv_elu_kernel<<<dim3((B_ * C_ * H_ * W_ + 255) / 256), dim3(256), 0, stream>>>(
      x, t_img, conv_w, conv_b, out);
}
